// KOrderGPMap_15264313770447
// MI455X (gfx1250) — compile-verified, matrix operand order checked by reading, not run on hardware
//
#include <hip/hip_runtime.h>

typedef __attribute__((ext_vector_type(16))) _Float16 v16h;
typedef __attribute__((ext_vector_type(8)))  float    v8f;

#define LL 64
#define CC 4
#define BB 32
#define PP 256           /* L*C */
#define NTILES 4096      /* (PP*PP)/16 row-tiles of theta3 viewed as 65536x256 */

// ---------------------------------------------------------------------------
// Order-3 term as fused WMMA GEMM + masked reduction.
//   T[m, r] = theta3[m*256 + r],  m=(u,a,v,c) flat, r=(w,e) flat
//   G[m, b] = sum_r T[m,r]*[u<v][v<w]*x[b,r]         (WMMA f32<-f16)
//   part[tile, b] = sum_{m in tile} G[m,b]*x[b,p(m)]*x[b,q(m)]
// ---------------------------------------------------------------------------
template <bool USE_ATOMIC>
__global__ __launch_bounds__(32) void korder3_wmma_kernel(
    const float* __restrict__ th3, const float* __restrict__ x,
    float* __restrict__ dst /* partials (tile-major) or out[] when atomic */) {
  const int tile   = blockIdx.x;
  const int m_base = tile << 4;
  const int p      = m_base >> 8;    // row-block index (u,a)  -- constant per tile
  const int q_base = m_base & 255;   // (v,c) base, multiple of 16 -> 4 v values
  const int u      = p >> 2;
  const int v_min  = q_base >> 2;

  const int lane = threadIdx.x;      // 0..31 (wave32)
  const int hi   = lane >> 4;
  const int colg = lane & 15;
  const int b0   = colg;             // batch for N-tile 0
  const int b1   = 16 + colg;        // batch for N-tile 1

  float p0 = 0.f, p1 = 0.f;

  if (u < v_min + 3) {               // tile has at least one row with v > u
    const int  row   = m_base + colg;          // A row (M = lane & 15)
    const int  vrow  = (row & 255) >> 2;       // v of this row
    const bool rowok = (u < vrow);
    const int  veff  = (v_min > u + 1) ? v_min : (u + 1);
    const float* rA  = th3 + ((size_t)row << 8);

    // warm the next tile's rows (global_prefetch_b8); streaming theta3 once.
    if (row + 16 < PP * PP) __builtin_prefetch(rA + (16 << 8), 0, 0);

    v8f acc0 = {}; v8f acc1 = {};

#pragma unroll
    for (int ks = 0; ks < 8; ++ks) {
      if (ks * 8 + 7 <= veff) continue;        // whole k-slab masked (w <= v)
      const int cb = ks * 32 + hi * 8;         // this lane's first K column

      // ---- A fragment: theta3 row, masked by (u<v)&&(v<w), f32 -> f16 ----
      const float4 a0 = *(const float4*)(rA + cb);
      const float4 a1 = *(const float4*)(rA + cb + 4);
      const float4 a2 = *(const float4*)(rA + cb + 16);
      const float4 a3 = *(const float4*)(rA + cb + 20);
      const float av[16] = {a0.x, a0.y, a0.z, a0.w, a1.x, a1.y, a1.z, a1.w,
                            a2.x, a2.y, a2.z, a2.w, a3.x, a3.y, a3.z, a3.w};
      v16h Af;
#pragma unroll
      for (int j = 0; j < 8; ++j) {
        const int  c1  = cb + j;
        const int  c2  = cb + 16 + j;
        const bool ok1 = rowok && (vrow < (c1 >> 2));
        const bool ok2 = rowok && (vrow < (c2 >> 2));
        Af[j]     = (_Float16)(ok1 ? av[j]     : 0.f);
        Af[8 + j] = (_Float16)(ok2 ? av[8 + j] : 0.f);
      }

      // ---- B fragments: one-hot x (exact in f16), two batch halves ----
      const float4 e0 = *(const float4*)(x + (size_t)b0 * PP + cb);
      const float4 e1 = *(const float4*)(x + (size_t)b0 * PP + cb + 4);
      const float4 e2 = *(const float4*)(x + (size_t)b0 * PP + cb + 16);
      const float4 e3 = *(const float4*)(x + (size_t)b0 * PP + cb + 20);
      const float4 f0 = *(const float4*)(x + (size_t)b1 * PP + cb);
      const float4 f1 = *(const float4*)(x + (size_t)b1 * PP + cb + 4);
      const float4 f2 = *(const float4*)(x + (size_t)b1 * PP + cb + 16);
      const float4 f3 = *(const float4*)(x + (size_t)b1 * PP + cb + 20);
      const float bv0[16] = {e0.x, e0.y, e0.z, e0.w, e1.x, e1.y, e1.z, e1.w,
                             e2.x, e2.y, e2.z, e2.w, e3.x, e3.y, e3.z, e3.w};
      const float bv1[16] = {f0.x, f0.y, f0.z, f0.w, f1.x, f1.y, f1.z, f1.w,
                             f2.x, f2.y, f2.z, f2.w, f3.x, f3.y, f3.z, f3.w};
      v16h Bf0, Bf1;
#pragma unroll
      for (int j = 0; j < 16; ++j) {
        Bf0[j] = (_Float16)bv0[j];
        Bf1[j] = (_Float16)bv1[j];
      }

      acc0 = __builtin_amdgcn_wmma_f32_16x16x32_f16(false, Af, false, Bf0,
                                                    (short)0, acc0, false, false);
      acc1 = __builtin_amdgcn_wmma_f32_16x16x32_f16(false, Af, false, Bf1,
                                                    (short)0, acc1, false, false);
    }

    // ---- fused stage 2: multiply by x[b,p]*x[b,q(m)] and reduce rows ----
    const float xp0 = x[(size_t)b0 * PP + p];
    const float xp1 = x[(size_t)b1 * PP + p];
    float s0 = 0.f, s1 = 0.f;
#pragma unroll
    for (int i = 0; i < 8; ++i) {          // D VGPR i holds row M = i + 8*hi
      const int q = q_base + hi * 8 + i;
      s0 += acc0[i] * x[(size_t)b0 * PP + q];
      s1 += acc1[i] * x[(size_t)b1 * PP + q];
    }
    p0 = xp0 * s0;
    p1 = xp1 * s1;
  }

  // combine row-halves (lanes l and l^16 hold the same batch column)
  p0 += __shfl_xor(p0, 16, 32);
  p1 += __shfl_xor(p1, 16, 32);

  if (lane < 16) {
    if (USE_ATOMIC) {
      if (p0 != 0.f) atomicAdd(&dst[b0], p0);
      if (p1 != 0.f) atomicAdd(&dst[b1], p1);
    } else {
      dst[(size_t)tile * BB + b0] = p0;   // must write even when skipped (ws poisoned)
      dst[(size_t)tile * BB + b1] = p1;
    }
  }
}

// ---------------------------------------------------------------------------
// Orders 0..2 (tiny: theta2 = 256KB) + deterministic fixed-order sum of the
// per-tile order-3 partials. One block per batch element.
// ---------------------------------------------------------------------------
__global__ __launch_bounds__(64) void korder_finalize_kernel(
    const float* __restrict__ x, const float* __restrict__ th0,
    const float* __restrict__ th1, const float* __restrict__ th2,
    const float* __restrict__ part, int ntiles, float* __restrict__ out) {
  __shared__ int   sel[LL];
  __shared__ float red[64];
  const int b = blockIdx.x;
  const int t = threadIdx.x;   // 0..63 == position u

  // decode one-hot selection a_u
  {
    int a = 0;
#pragma unroll
    for (int c = 1; c < CC; ++c)
      if (x[(size_t)b * PP + t * CC + c] > 0.5f) a = c;
    sel[t] = a;
  }
  __syncthreads();

  float s = th1[t * CC + sel[t]];                       // order 1

  for (int i = t; i < LL * LL; i += 64) {               // order 2, u < v
    const int uu = i >> 6, vv = i & (LL - 1);
    if (uu < vv) s += th2[(size_t)(uu * CC + sel[uu]) * PP + vv * CC + sel[vv]];
  }

  if (part)                                             // order-3 partials
    for (int i = t; i < ntiles; i += 64) s += part[(size_t)i * BB + b];

  red[t] = s;
  __syncthreads();
#pragma unroll
  for (int off = 32; off > 0; off >>= 1) {
    if (t < off) red[t] += red[t + off];
    __syncthreads();
  }
  if (t == 0) out[b] = (part ? th0[0] : th0[0]) + red[0];
}

extern "C" void kernel_launch(void* const* d_in, const int* in_sizes, int n_in,
                              void* d_out, int out_size, void* d_ws, size_t ws_size,
                              hipStream_t stream) {
  (void)in_sizes; (void)n_in; (void)out_size;
  const float* x   = (const float*)d_in[0];
  const float* th0 = (const float*)d_in[1];
  const float* th1 = (const float*)d_in[2];
  const float* th2 = (const float*)d_in[3];
  const float* th3 = (const float*)d_in[4];
  float* out = (float*)d_out;

  const size_t PART_BYTES = (size_t)NTILES * BB * sizeof(float);
  if (ws_size >= PART_BYTES) {
    // deterministic two-pass path
    float* part = (float*)d_ws;
    korder3_wmma_kernel<false><<<NTILES, 32, 0, stream>>>(th3, x, part);
    korder_finalize_kernel<<<BB, 64, 0, stream>>>(x, th0, th1, th2, part, NTILES, out);
  } else {
    // fallback: init out with orders 0..2, then atomic-accumulate order 3
    korder_finalize_kernel<<<BB, 64, 0, stream>>>(x, th0, th1, th2, nullptr, 0, out);
    korder3_wmma_kernel<true><<<NTILES, 32, 0, stream>>>(th3, x, out);
  }
}
